// PraxisPEER_54125177864378
// MI455X (gfx1250) — matrix operand order repaired and checked by reading, not compiled, not run on hardware
//
#include <hip/hip_runtime.h>
#include <hip/hip_bf16.h>
#include <math.h>

// ---------------------------------------------------------------------------
// PEER (product-key experts) for MI455X / gfx1250.
// Roofline: expert gather = ~1.07 GB HBM traffic (dominant, ~46us @23.3TB/s);
// GEMMs (8.6 + 2.15 GFLOP) run on bf16 WMMA (v_wmma_f32_16x16x32_bf16),
// ping-pong double-buffered (no register rotation copies -> no spills).
// ---------------------------------------------------------------------------

typedef __attribute__((ext_vector_type(16))) __bf16        v16bf;
typedef __attribute__((ext_vector_type(8)))  float         v8f;
typedef __attribute__((ext_vector_type(4)))  unsigned int  v4u;

union Frag { v16bf v; v4u u[2]; };

static __device__ inline unsigned short f2bf(float x) {
  unsigned int u = __float_as_uint(x);
  u += 0x7FFFu + ((u >> 16) & 1u);      // round-to-nearest-even
  return (unsigned short)(u >> 16);
}

// A-matrix 16x32 bf16 fragment (ISA 7.12.2): lane l -> row l%16,
// V0-3: K = (l/16)*8 .. +7 ; V4-7: K = 16 + (l/16)*8 .. +7
static __device__ inline v16bf load_frag_a(const unsigned short* __restrict__ row,
                                           int kb, int lane) {
  const int c = (lane >> 4) * 8;
  Frag f;
  f.u[0] = *(const v4u*)(row + kb + c);
  f.u[1] = *(const v4u*)(row + kb + 16 + c);
  return f.v;
}

// B-matrix 32x16 bf16 fragment: lane l -> column l%16,
// V0-3: K = (l/16)*8 .. +7 ; V4-7: K = 16 + (l/16)*8 .. +7 (same as A).
static __device__ inline v16bf load_frag_b(const unsigned short* __restrict__ col,
                                           int kb, int lane) {
  const unsigned short* p = col + kb + (lane >> 4) * 8;
  Frag f;
  f.u[0] = *(const v4u*)(p);
  f.u[1] = *(const v4u*)(p + 16);
  return f.v;
}

static __device__ inline v8f wmma_bf16(v16bf a, v16bf b, v8f c) {
  return __builtin_amdgcn_wmma_f32_16x16x32_bf16(false, a, false, b,
                                                 (short)0, c, false, false);
}

// ------------------------- constants ---------------------------------------
#define DD   1024     // model dim
#define HH   8        // heads
#define KD   128      // key dim
#define NK   256      // keys per half
#define QN   2048     // 2*HH*KD  (w_q rows)
#define TOPK 8

// ------------------------- BatchNorm stats ---------------------------------
__global__ void k_bn_partial(const float* __restrict__ x, float* __restrict__ sums,
                             float* __restrict__ sumsq, int rows_per_chunk) {
  const int f = blockIdx.x * blockDim.x + threadIdx.x;     // feature
  const int r0 = blockIdx.y * rows_per_chunk;
  float s = 0.f, s2 = 0.f;
  for (int r = 0; r < rows_per_chunk; ++r) {
    float v = x[(size_t)(r0 + r) * DD + f];
    s += v; s2 += v * v;
  }
  atomicAdd(&sums[f], s);
  atomicAdd(&sumsq[f], s2);
}

__global__ void k_bn_finalize(const float* __restrict__ sums,
                              const float* __restrict__ sumsq,
                              float* __restrict__ mu, float* __restrict__ rs,
                              float inv_n) {
  const int f = blockIdx.x * blockDim.x + threadIdx.x;
  float m = sums[f] * inv_n;
  float var = sumsq[f] * inv_n - m * m;
  mu[f] = m;
  rs[f] = rsqrtf(var + 1e-5f);
}

__global__ void k_bn_apply(const float* __restrict__ x, const float* __restrict__ mu,
                           const float* __restrict__ rs, const float* __restrict__ gamma,
                           const float* __restrict__ beta, unsigned short* __restrict__ xn) {
  const int i = blockIdx.x * blockDim.x + threadIdx.x;
  const int f = i & (DD - 1);
  float v = (x[i] - mu[f]) * rs[f] * gamma[f] + beta[f];
  xn[i] = f2bf(v);
}

__global__ void k_cast_bf16(const float* __restrict__ in, unsigned short* __restrict__ out) {
  const int i = blockIdx.x * blockDim.x + threadIdx.x;
  out[i] = f2bf(in[i]);
}

// ------------------------- Q = XN * Wq^T (bf16 WMMA, NT) --------------------
// A: [M x 1024] bf16 row-major, B: [2048 x 1024] bf16 row-major (rows = out col)
// C: [M x 2048] bf16.  One wave -> 16x64 tile (4 accumulators, A reused).
// Ping-pong double buffer (K step 64 = 2 WMMA steps); buffers alternate by
// being load targets, never copied -> no spills, loads overlap WMMA.
__global__ __launch_bounds__(256, 1) void k_gemm_q(
    const unsigned short* __restrict__ A,
    const unsigned short* __restrict__ B,
    unsigned short* __restrict__ C) {
  const int lane = threadIdx.x & 31;
  const int wave = threadIdx.x >> 5;
  const int tile = blockIdx.x * 8 + wave;       // wave-tile id
  const int mt = tile >> 5;                     // / (2048/64)
  const int nt = tile & 31;
  v8f acc[4] = {};
  const unsigned short* Arow = A + (size_t)(mt * 16 + (lane & 15)) * DD;
  const unsigned short* Bcol[4];
#pragma unroll
  for (int j = 0; j < 4; ++j)
    Bcol[j] = B + (size_t)(nt * 64 + j * 16 + (lane & 15)) * DD;

  v16bf aA = load_frag_a(Arow, 0, lane);
  v16bf bA[4];
#pragma unroll
  for (int j = 0; j < 4; ++j) bA[j] = load_frag_b(Bcol[j], 0, lane);

  for (int kb = 0; kb < DD - 64; kb += 64) {
    v16bf aB = load_frag_a(Arow, kb + 32, lane);
    v16bf bB[4];
#pragma unroll
    for (int j = 0; j < 4; ++j) bB[j] = load_frag_b(Bcol[j], kb + 32, lane);
#pragma unroll
    for (int j = 0; j < 4; ++j) acc[j] = wmma_bf16(aA, bA[j], acc[j]);

    aA = load_frag_a(Arow, kb + 64, lane);
#pragma unroll
    for (int j = 0; j < 4; ++j) bA[j] = load_frag_b(Bcol[j], kb + 64, lane);
#pragma unroll
    for (int j = 0; j < 4; ++j) acc[j] = wmma_bf16(aB, bB[j], acc[j]);
  }
  {   // tail: last two K-steps (kb = DD-64, DD-32)
    v16bf aB = load_frag_a(Arow, DD - 32, lane);
    v16bf bB[4];
#pragma unroll
    for (int j = 0; j < 4; ++j) bB[j] = load_frag_b(Bcol[j], DD - 32, lane);
#pragma unroll
    for (int j = 0; j < 4; ++j) acc[j] = wmma_bf16(aA, bA[j], acc[j]);
#pragma unroll
    for (int j = 0; j < 4; ++j) acc[j] = wmma_bf16(aB, bB[j], acc[j]);
  }

  const int rowbase = mt * 16 + 8 * (lane >> 4);
#pragma unroll
  for (int j = 0; j < 4; ++j) {
    const int col = nt * 64 + j * 16 + (lane & 15);
#pragma unroll
    for (int r = 0; r < 8; ++r)
      C[(size_t)(rowbase + r) * QN + col] = f2bf(acc[j][r]);
  }
}

// ------------------------- sim = Q_ph * keys_ph^T ---------------------------
// For each (p,head): M=T tokens, N=256, K=128.  sim layout [2][T][8][256] f32.
// 2-deep pipeline, fully unrolled (rotation is SSA-renamed, no real moves).
__global__ __launch_bounds__(256, 1) void k_sim(
    const unsigned short* __restrict__ Q,
    const unsigned short* __restrict__ KY,
    float* __restrict__ sim, int T) {
  const int lane = threadIdx.x & 31;
  const int wave = threadIdx.x >> 5;
  const int tt = blockIdx.x;                       // token tile
  const int wtile = blockIdx.y * 8 + wave;         // 0..63
  const int ph = wtile >> 2;
  const int p = ph >> 3, head = ph & 7;
  const int nt = wtile & 3;                        // 64-col tile in 256
  v8f acc[4] = {};
  const unsigned short* Arow =
      Q + (size_t)(tt * 16 + (lane & 15)) * QN + p * (HH * KD) + head * KD;
  const unsigned short* Bcol[4];
#pragma unroll
  for (int j = 0; j < 4; ++j)
    Bcol[j] = KY + (size_t)((head * NK + nt * 64 + j * 16 + (lane & 15)) * 2 + p) * KD;

  v16bf aP = load_frag_a(Arow, 0, lane);
  v16bf bP[4];
#pragma unroll
  for (int j = 0; j < 4; ++j) bP[j] = load_frag_b(Bcol[j], 0, lane);
#pragma unroll
  for (int s = 0; s < 4; ++s) {             // fully unrolled: renames, not copies
    v16bf aN; v16bf bN[4];
    if (s < 3) {
      aN = load_frag_a(Arow, (s + 1) * 32, lane);
#pragma unroll
      for (int j = 0; j < 4; ++j) bN[j] = load_frag_b(Bcol[j], (s + 1) * 32, lane);
    }
#pragma unroll
    for (int j = 0; j < 4; ++j) acc[j] = wmma_bf16(aP, bP[j], acc[j]);
    aP = aN;
#pragma unroll
    for (int j = 0; j < 4; ++j) bP[j] = bN[j];
  }

  const int half = lane >> 4;
#pragma unroll
  for (int j = 0; j < 4; ++j) {
    const int n = nt * 64 + j * 16 + (lane & 15);
#pragma unroll
    for (int r = 0; r < 8; ++r) {
      const int t = tt * 16 + r + 8 * half;
      sim[(((size_t)p * T + t) * HH + head) * NK + n] = acc[j][r];
    }
  }
}

// ------------------------- two-stage top-k + softmax ------------------------
static __device__ void topk8_256(const float* __restrict__ base, int lane,
                                 float* outv, int* outi) {
  float v[8]; int idx[8]; int mask = 0;
#pragma unroll
  for (int i = 0; i < 8; ++i) { int j = lane + 32 * i; v[i] = base[j]; idx[i] = j; }
  for (int r = 0; r < TOPK; ++r) {
    float bm = -3.402823466e38f; int bi = 0x7fffffff; int bslot = -1;
#pragma unroll
    for (int i = 0; i < 8; ++i)
      if (!((mask >> i) & 1) && (v[i] > bm || (v[i] == bm && idx[i] < bi))) {
        bm = v[i]; bi = idx[i]; bslot = i;
      }
    float m = bm; int mi = bi;
    for (int off = 16; off; off >>= 1) {
      float om = __shfl_xor(m, off, 32);
      int   oi = __shfl_xor(mi, off, 32);
      if (om > m || (om == m && oi < mi)) { m = om; mi = oi; }
    }
    outv[r] = m; outi[r] = mi;
    if (bi == mi && bslot >= 0) mask |= (1 << bslot);
  }
}

__global__ void k_topk(const float* __restrict__ sim, int* __restrict__ oidx,
                       float* __restrict__ ogate, int T) {
  const int lane = threadIdx.x & 31;
  const int wave = threadIdx.x >> 5;
  const int item = blockIdx.x * 8 + wave;      // (token, head)
  const int t = item >> 3, head = item & 7;

  float sx[8], sy[8]; int ix8[8], iy8[8];
  topk8_256(sim + (((size_t)0 * T + t) * HH + head) * NK, lane, sx, ix8);
  topk8_256(sim + (((size_t)1 * T + t) * HH + head) * NK, lane, sy, iy8);

  // stage 2: 64 combos, 2 per lane; tie-break on combo position (jax order)
  float cv[2]; int cpos[2]; int cidx[2]; int mask = 0;
#pragma unroll
  for (int i = 0; i < 2; ++i) {
    const int c = lane + 32 * i;
    cv[i]   = sx[c >> 3] + sy[c & 7];
    cpos[i] = c;
    cidx[i] = ix8[c >> 3] * NK + iy8[c & 7];
  }
  float s[8]; int si[8];
  for (int r = 0; r < TOPK; ++r) {
    float bm = -3.402823466e38f; int bp = 0x7fffffff; int bx = 0; int bslot = -1;
#pragma unroll
    for (int i = 0; i < 2; ++i)
      if (!((mask >> i) & 1) && (cv[i] > bm || (cv[i] == bm && cpos[i] < bp))) {
        bm = cv[i]; bp = cpos[i]; bx = cidx[i]; bslot = i;
      }
    float m = bm; int mp = bp; int mx = bx;
    for (int off = 16; off; off >>= 1) {
      float om = __shfl_xor(m, off, 32);
      int   op = __shfl_xor(mp, off, 32);
      int   ox = __shfl_xor(mx, off, 32);
      if (om > m || (om == m && op < mp)) { m = om; mp = op; mx = ox; }
    }
    s[r] = m; si[r] = mx;
    if (bp == mp && bslot >= 0) mask |= (1 << bslot);
  }
  // softmax over 8 selected scores (all lanes hold them)
  float mx = s[0];
#pragma unroll
  for (int r = 1; r < 8; ++r) mx = fmaxf(mx, s[r]);
  float e[8]; float sum = 0.f;
#pragma unroll
  for (int r = 0; r < 8; ++r) { e[r] = __expf(s[r] - mx); sum += e[r]; }
  if (lane < 8) {
    oidx[(size_t)item * 8 + lane]  = si[lane];
    ogate[(size_t)item * 8 + lane] = e[lane] / sum;
  }
}

// ------------------------- expert gather / apply ----------------------------
// One block per token.  Phase 1: 64 down-projection dots (8 waves x 8 experts),
// exact GELU * softmax gate -> coefficients in LDS.  Phase 2: each thread owns
// 4 output features and streams 64 coalesced key_out rows (HBM-bound phase).
__global__ void k_expert(const float* __restrict__ x, const float* __restrict__ win,
                         const float* __restrict__ wout, const int* __restrict__ idx,
                         const float* __restrict__ gate, float* __restrict__ out) {
  __shared__ float xs[DD];
  __shared__ float coef[64];
  __shared__ int   eidx[64];
  const int t = blockIdx.x;
  const int tid = threadIdx.x;
  ((float4*)xs)[tid] = ((const float4*)(x + (size_t)t * DD))[tid];
  if (tid < 64) eidx[tid] = idx[(size_t)t * 64 + tid];
  __syncthreads();

  const int lane = tid & 31, wave = tid >> 5;
  for (int q = 0; q < 8; ++q) {
    const int e = wave * 8 + q;
    const float4* w = (const float4*)(win + (size_t)eidx[e] * DD);
    float acc = 0.f;
#pragma unroll
    for (int i = 0; i < 8; ++i) {
      float4 a = w[lane + 32 * i];
      float4 b = ((const float4*)xs)[lane + 32 * i];
      acc += a.x * b.x + a.y * b.y + a.z * b.z + a.w * b.w;
    }
    for (int off = 16; off; off >>= 1) acc += __shfl_xor(acc, off, 32);
    if (lane == 0) {
      float g  = gate[(size_t)t * 64 + e];
      float ge = 0.5f * acc * (1.0f + erff(acc * 0.70710678118654752f));  // exact GELU
      coef[e] = ge * g;
    }
  }
  __syncthreads();

  float4 acc = {0.f, 0.f, 0.f, 0.f};
#pragma unroll 4
  for (int e = 0; e < 64; ++e) {
    // prefetch the row we'll need 8 experts from now (global_prefetch_b8)
    if (e + 8 < 64)
      __builtin_prefetch(wout + (size_t)eidx[e + 8] * DD + tid * 4, 0, 1);
    const float c = coef[e];
    float4 w = ((const float4*)(wout + (size_t)eidx[e] * DD))[tid];
    acc.x += c * w.x; acc.y += c * w.y; acc.z += c * w.z; acc.w += c * w.w;
  }
  ((float4*)(out + (size_t)t * DD))[tid] = acc;
}

// ------------------------- host launcher ------------------------------------
extern "C" void kernel_launch(void* const* d_in, const int* in_sizes, int n_in,
                              void* d_out, int out_size, void* d_ws, size_t ws_size,
                              hipStream_t stream) {
  const float* inp   = (const float*)d_in[0];   // (b, n, d)
  const float* gamma = (const float*)d_in[1];
  const float* beta  = (const float*)d_in[2];
  const float* wq    = (const float*)d_in[3];   // (2048, 1024)
  const float* keys  = (const float*)d_in[4];   // (8, 256, 2, 128)
  const float* kin   = (const float*)d_in[5];   // (65536, 1024)
  const float* kout  = (const float*)d_in[6];   // (65536, 1024)
  float* out = (float*)d_out;

  const int T = in_sizes[0] / DD;               // 2048 tokens

  // workspace layout
  char* w = (char*)d_ws;
  float* sums  = (float*)w;                 w += DD * sizeof(float);
  float* sumsq = (float*)w;                 w += DD * sizeof(float);
  float* mu    = (float*)w;                 w += DD * sizeof(float);
  float* rs    = (float*)w;                 w += DD * sizeof(float);
  unsigned short* xn    = (unsigned short*)w; w += (size_t)T * DD * 2;
  unsigned short* wqb   = (unsigned short*)w; w += (size_t)QN * DD * 2;
  unsigned short* keysb = (unsigned short*)w; w += (size_t)HH * NK * 2 * KD * 2;
  unsigned short* qb    = (unsigned short*)w; w += (size_t)T * QN * 2;
  float* sim   = (float*)w;                 w += (size_t)2 * T * HH * NK * sizeof(float);
  int*   eidx  = (int*)w;                   w += (size_t)T * 64 * sizeof(int);
  float* gate  = (float*)w;                 w += (size_t)T * 64 * sizeof(float);

  // 1) BatchNorm statistics
  hipMemsetAsync(sums, 0, 2 * DD * sizeof(float), stream);
  const int chunks = 16;
  k_bn_partial<<<dim3(DD / 256, chunks), 256, 0, stream>>>(inp, sums, sumsq, T / chunks);
  k_bn_finalize<<<DD / 256, 256, 0, stream>>>(sums, sumsq, mu, rs, 1.0f / (float)T);

  // 2) normalize + casts to bf16
  k_bn_apply<<<(T * DD) / 256, 256, 0, stream>>>(inp, mu, rs, gamma, beta, xn);
  k_cast_bf16<<<(QN * DD) / 256, 256, 0, stream>>>(wq, wqb);
  k_cast_bf16<<<(HH * NK * 2 * KD) / 256, 256, 0, stream>>>(keys, keysb);

  // 3) Q = XN * Wq^T   (WMMA bf16)
  k_gemm_q<<<(T / 16) * (QN / 64) / 8, 256, 0, stream>>>(xn, wqb, qb);

  // 4) sim = Q_ph * keys_ph^T  (WMMA bf16)
  k_sim<<<dim3(T / 16, 8), 256, 0, stream>>>(qb, keysb, sim, T);

  // 5) two-stage top-k + softmax gates
  k_topk<<<T, 256, 0, stream>>>(sim, eidx, gate, T);

  // 6) expert gather / GELU / weighted sum (HBM-bound phase)
  k_expert<<<T, 256, 0, stream>>>(inp, kin, kout, eidx, gate, out);
}
